// GCNModel_83562883711392
// MI455X (gfx1250) — matrix operand order repaired and checked by reading, not compiled, not run on hardware
//
#include <hip/hip_runtime.h>

// ---------------------------------------------------------------------------
// GCN forward: h = relu( (adj_norm @ h) @ W + b ) x3, 32 graphs, N=1024.
// adj normalized once -> bf16 (64MB, L2-resident), all GEMMs via
// v_wmma_f32_16x16x32_bf16 (wave32), async global->LDS double-buffered
// staging (ASYNCcnt), bias+relu fused in epilogue. Intermediate activations
// are kept TRANSPOSED in global memory so every LDS stage is a plain 2D
// b128 async copy (no in-flight transpose needed), and transposed epilogue
// stores are single b128 per lane.
// ---------------------------------------------------------------------------

typedef __attribute__((ext_vector_type(16))) __bf16 bv16;
typedef __attribute__((ext_vector_type(8)))  float v8f;
typedef int v4i __attribute__((vector_size(4 * sizeof(int))));

#define NG    32      // B*W graphs
#define NN    1024    // nodes per graph
#define FIN   256
#define FHID  256
#define FOUT  128

#define BM 128
#define BN 128
#define BK 32
#define LDT 40        // padded LDS stride (bf16 units): 80B rows, 16B aligned

#if defined(__has_builtin)
#if __has_builtin(__builtin_amdgcn_global_load_async_to_lds_b128)
#define USE_ASYNC_COPY 1
#endif
#endif

__device__ __forceinline__ unsigned short f2bf(float f) {
  unsigned int u = __float_as_uint(f);
  unsigned int r = (u + 0x7FFFu + ((u >> 16) & 1u)) >> 16;  // RNE
  return (unsigned short)r;
}

// 16-byte global -> LDS copy; async (ASYNCcnt) when the toolchain has it.
__device__ __forceinline__ void copy16_g2l(const unsigned short* g, unsigned short* l) {
#ifdef USE_ASYNC_COPY
  // signature (probe-confirmed via diagnostic): (v4i AS1*, v4i AS3*, imm, imm)
  __builtin_amdgcn_global_load_async_to_lds_b128(
      (__attribute__((address_space(1))) v4i*)(unsigned long long)(uintptr_t)g,
      (__attribute__((address_space(3))) v4i*)(unsigned int)(uintptr_t)l,
      0, 0);
#else
  *(uint4*)l = *(const uint4*)g;
#endif
}

template <int N>
__device__ __forceinline__ void wait_async() {
#ifdef USE_ASYNC_COPY
#if __has_builtin(__builtin_amdgcn_s_wait_asynccnt)
  __builtin_amdgcn_s_wait_asynccnt(N);
#else
  asm volatile("s_wait_asynccnt %0" ::"i"(N) : "memory");
#endif
#endif
}

// ---------------------------------------------------------------------------
// deg[g*N+i] = rsqrt( sum_j adj[g,i,j] + 1 )   (self-loop adds 1)
// ---------------------------------------------------------------------------
__global__ __launch_bounds__(256) void degree_kernel(
    const float* __restrict__ adj, float* __restrict__ deg) {
  __shared__ float red[256];
  const int row = blockIdx.x;                  // g*NN + i
  const int t = threadIdx.x;
  const float4 v = ((const float4*)(adj + (size_t)row * NN))[t];
  red[t] = v.x + v.y + v.z + v.w;
  __syncthreads();
  #pragma unroll
  for (int off = 128; off > 0; off >>= 1) {
    if (t < off) red[t] += red[t + off];
    __syncthreads();
  }
  if (t == 0) deg[row] = rsqrtf(red[0] + 1.0f);
}

// ---------------------------------------------------------------------------
// adjn[g,i,j] = bf16( (adj[g,i,j] + (i==j)) * d_i * d_j )
// ---------------------------------------------------------------------------
__global__ __launch_bounds__(256) void normalize_kernel(
    const float* __restrict__ adj, const float* __restrict__ deg,
    unsigned short* __restrict__ out) {
  const size_t quad = (size_t)blockIdx.x * 256 + threadIdx.x;
  const size_t e = quad * 4;
  const int g   = (int)(e >> 20);              // NN*NN = 2^20
  const int rem = (int)(e & ((1u << 20) - 1u));
  const int i   = rem >> 10;
  const int j0  = rem & 1023;
  const float di = deg[g * NN + i];
  const float* __restrict__ dj = deg + g * NN;
  const float4 v = *(const float4*)(adj + e);
  float f[4] = {v.x, v.y, v.z, v.w};
  unsigned short o[4];
  #pragma unroll
  for (int k = 0; k < 4; ++k) {
    const int j = j0 + k;
    float val = f[k] + (i == j ? 1.0f : 0.0f);
    o[k] = f2bf(val * di * dj[j]);
  }
  uint2 p;
  p.x = (unsigned)o[0] | ((unsigned)o[1] << 16);
  p.y = (unsigned)o[2] | ((unsigned)o[3] << 16);
  *(uint2*)(out + e) = p;
}

// ---------------------------------------------------------------------------
// Batched transpose + f32->bf16:  in[g] (RxC f32) -> out[g] (CxR bf16)
// block 256 = 32x8, 32x32 LDS tile; grid (C/32, R/32, G)
// ---------------------------------------------------------------------------
__global__ __launch_bounds__(256) void tcvt_kernel(
    const float* __restrict__ in, unsigned short* __restrict__ out,
    int R, int C) {
  __shared__ float tile[32][33];
  const int g  = blockIdx.z;
  const int r0 = blockIdx.y * 32;
  const int c0 = blockIdx.x * 32;
  in  += (size_t)g * R * C;
  out += (size_t)g * R * C;
  const int tx = threadIdx.x & 31;
  const int ty = threadIdx.x >> 5;             // 0..7
  #pragma unroll
  for (int i = 0; i < 4; ++i) {
    const int r = ty + i * 8;
    tile[r][tx] = in[(size_t)(r0 + r) * C + c0 + tx];
  }
  __syncthreads();
  #pragma unroll
  for (int i = 0; i < 4; ++i) {
    const int c = ty + i * 8;
    out[(size_t)(c0 + c) * R + r0 + tx] = f2bf(tile[tx][c]);
  }
}

// ---------------------------------------------------------------------------
// Tiled bf16 GEMM, f32 accumulation:  C[g] = op( A[g] (MxK) @ Bt[g]^T )
//   A  row-major MxK, Bt row-major NcxK (i.e. B transposed)
//   OUTMODE: 0 = bf16 row-major, 1 = bf16 transposed (Nc x M), 2 = f32 row-major
//   EPI: fuse +bias[col], relu
// Block tile 128x128, K-step 32, 8 wave32 waves in 4(M)x2(N); each wave owns
// 32x64 = 2x4 WMMA 16x16x32 accumulators. Double-buffered LDS fed by async
// global->LDS b128 copies (4 per thread per K-step), overlapped with WMMA.
// grid = (Nc/128, M/128, NG)
// ---------------------------------------------------------------------------
template <int OUTMODE, bool EPI>
__global__ __launch_bounds__(256) void gemm_bf16_kernel(
    const unsigned short* __restrict__ A, size_t strideA, int lda,
    const unsigned short* __restrict__ Bt, size_t strideB, int ldb,
    const float* __restrict__ bias,
    unsigned short* __restrict__ Cb, float* __restrict__ Cf,
    size_t strideC, int ldc, int K) {
  __shared__ unsigned short ldsA[2][BM * LDT];
  __shared__ unsigned short ldsB[2][BN * LDT];

  const int g     = blockIdx.z;
  const int mBase = blockIdx.y * BM;
  const int nBase = blockIdx.x * BN;
  A  += (size_t)g * strideA + (size_t)mBase * lda;
  Bt += (size_t)g * strideB + (size_t)nBase * ldb;

  const int t    = threadIdx.x;
  const int lane = t & 31;
  const int wave = t >> 5;
  const int wm   = (wave & 3) * 32;   // wave M offset in block tile
  const int wn   = (wave >> 2) * 64;  // wave N offset in block tile
  const int r15  = lane & 15;
  const int hi   = lane >> 4;

  // staging map: 128 rows x 4 16B-chunks per matrix; thread does rows {sr, sr+64}
  const int sr = t >> 2;              // 0..63
  const int sc = (t & 3) << 3;        // 0,8,16,24 (bf16 units)

  auto stage = [&](int buf, int kBase) {
    #pragma unroll
    for (int i = 0; i < 2; ++i) {
      const int r = sr + i * 64;
      copy16_g2l(A + (size_t)r * lda + kBase + sc, &ldsA[buf][r * LDT + sc]);
    }
    #pragma unroll
    for (int i = 0; i < 2; ++i) {
      const int r = sr + i * 64;
      copy16_g2l(Bt + (size_t)r * ldb + kBase + sc, &ldsB[buf][r * LDT + sc]);
    }
  };

  v8f zacc = {};
  v8f acc[2][4];
  #pragma unroll
  for (int i = 0; i < 2; ++i)
    #pragma unroll
    for (int j = 0; j < 4; ++j) acc[i][j] = zacc;

  union ABfrag { uint4 u[2]; bv16 v; };

  const int ktiles = K >> 5;
  stage(0, 0);
  for (int kt = 0; kt < ktiles; ++kt) {
    const int buf = kt & 1;
    if (kt + 1 < ktiles) {
      stage(buf ^ 1, (kt + 1) << 5);   // overlap next tile with compute
      wait_async<4>();                 // current tile landed; 4 newest in flight
    } else {
      wait_async<0>();
    }
    __syncthreads();

    const unsigned short* lA = ldsA[buf];
    const unsigned short* lB = ldsB[buf];

    // A 16x32 frag: lanes 0-15 hold K{0-7,16-23} of row r15; lanes 16-31 K{8-15,24-31}
    bv16 afrag[2];
    #pragma unroll
    for (int mi = 0; mi < 2; ++mi) {
      ABfrag tmp;
      const unsigned short* p = lA + (wm + mi * 16 + r15) * LDT + hi * 8;
      tmp.u[0] = *(const uint4*)(p);
      tmp.u[1] = *(const uint4*)(p + 16);
      afrag[mi] = tmp.v;
    }
    // B 32x16 frag: lanes 0-15 hold K{0-7,16-23} of col r15; lanes 16-31 K{8-15,24-31}
    bv16 bfrag[4];
    #pragma unroll
    for (int ni = 0; ni < 4; ++ni) {
      ABfrag tmp;
      const unsigned short* p = lB + (wn + ni * 16 + r15) * LDT + hi * 8;
      tmp.u[0] = *(const uint4*)(p);
      tmp.u[1] = *(const uint4*)(p + 16);
      bfrag[ni] = tmp.v;
    }

    #pragma unroll
    for (int mi = 0; mi < 2; ++mi)
      #pragma unroll
      for (int ni = 0; ni < 4; ++ni)
        acc[mi][ni] = __builtin_amdgcn_wmma_f32_16x16x32_bf16(
            false, afrag[mi], false, bfrag[ni], (short)0, acc[mi][ni],
            false, false);

    __syncthreads();  // protect buf before it is re-staged
  }

  // ---- epilogue: lane holds col r15, rows {r + 8*hi} of each 16x16 tile ----
  #pragma unroll
  for (int mi = 0; mi < 2; ++mi) {
    const int rowb = mBase + wm + mi * 16 + hi * 8;
    #pragma unroll
    for (int ni = 0; ni < 4; ++ni) {
      const int col = nBase + wn + ni * 16 + r15;
      const float bval = EPI ? bias[col] : 0.0f;
      if (OUTMODE == 1) {
        // transposed bf16: 8 consecutive rows of one col -> single b128 store
        union { uint4 u; unsigned short s[8]; } pk;
        #pragma unroll
        for (int r = 0; r < 8; ++r) {
          float val = acc[mi][ni][r];
          if (EPI) val = fmaxf(val + bval, 0.0f);
          pk.s[r] = f2bf(val);
        }
        *(uint4*)(Cb + (size_t)g * strideC + (size_t)col * ldc + rowb) = pk.u;
      } else {
        #pragma unroll
        for (int r = 0; r < 8; ++r) {
          float val = acc[mi][ni][r];
          if (EPI) val = fmaxf(val + bval, 0.0f);
          const size_t off = (size_t)g * strideC + (size_t)(rowb + r) * ldc + col;
          if (OUTMODE == 2) Cf[off] = val;
          else              Cb[off] = f2bf(val);
        }
      }
    }
  }
}

// ---------------------------------------------------------------------------
extern "C" void kernel_launch(void* const* d_in, const int* in_sizes, int n_in,
                              void* d_out, int out_size, void* d_ws, size_t ws_size,
                              hipStream_t stream) {
  (void)in_sizes; (void)n_in; (void)out_size; (void)ws_size;

  const float* X   = (const float*)d_in[0];
  const float* adj = (const float*)d_in[1];
  const float* W1  = (const float*)d_in[2];
  const float* b1  = (const float*)d_in[3];
  const float* W2  = (const float*)d_in[4];
  const float* b2  = (const float*)d_in[5];
  const float* W3  = (const float*)d_in[6];
  const float* b3  = (const float*)d_in[7];
  float* out = (float*)d_out;

  // workspace layout
  char* ws = (char*)d_ws;
  size_t off = 0;
  unsigned short* adjn = (unsigned short*)(ws + off); off += (size_t)NG * NN * NN * 2;   // 64MB
  unsigned short* Ht0  = (unsigned short*)(ws + off); off += (size_t)NG * FHID * NN * 2; // 16MB [g][f][n]
  unsigned short* Ht1  = (unsigned short*)(ws + off); off += (size_t)NG * FHID * NN * 2; // 16MB
  unsigned short* Y    = (unsigned short*)(ws + off); off += (size_t)NG * NN * FHID * 2; // 16MB [g][n][f]
  float*          deg  = (float*)(ws + off);          off += (size_t)NG * NN * 4;
  unsigned short* Wt1  = (unsigned short*)(ws + off); off += (size_t)FIN * FHID * 2;     // [o][f]
  unsigned short* Wt2  = (unsigned short*)(ws + off); off += (size_t)FHID * FHID * 2;
  unsigned short* Wt3  = (unsigned short*)(ws + off); off += (size_t)FHID * FOUT * 2;

  // 1) degrees + symmetric normalization into bf16 (once, reused x3)
  degree_kernel<<<NG * NN, 256, 0, stream>>>(adj, deg);
  normalize_kernel<<<((size_t)NG * NN * NN / 4) / 256, 256, 0, stream>>>(adj, deg, adjn);

  // 2) transpose+convert inputs/weights:  X -> Ht0 [g][f][n],  W -> Wt [o][f]
  tcvt_kernel<<<dim3(FIN / 32, NN / 32, NG), 256, 0, stream>>>(X, Ht0, NN, FIN);
  tcvt_kernel<<<dim3(FHID / 32, FIN / 32, 1), 256, 0, stream>>>(W1, Wt1, FIN, FHID);
  tcvt_kernel<<<dim3(FHID / 32, FHID / 32, 1), 256, 0, stream>>>(W2, Wt2, FHID, FHID);
  tcvt_kernel<<<dim3(FOUT / 32, FHID / 32, 1), 256, 0, stream>>>(W3, Wt3, FHID, FOUT);

  const size_t sAdj = (size_t)NN * NN;
  const size_t sH   = (size_t)NN * FHID;

  // ---- layer 1:  Y = adjn @ H0 ;  Ht1 = (relu(Y@W1+b1))^T ----
  gemm_bf16_kernel<0, false><<<dim3(FIN / BN, NN / BM, NG), 256, 0, stream>>>(
      adjn, sAdj, NN, Ht0, sH, NN, nullptr, Y, nullptr, sH, FIN, NN);
  gemm_bf16_kernel<1, true><<<dim3(FHID / BN, NN / BM, NG), 256, 0, stream>>>(
      Y, sH, FIN, Wt1, 0, FIN, b1, Ht1, nullptr, sH, NN, FIN);

  // ---- layer 2 ----
  gemm_bf16_kernel<0, false><<<dim3(FHID / BN, NN / BM, NG), 256, 0, stream>>>(
      adjn, sAdj, NN, Ht1, sH, NN, nullptr, Y, nullptr, sH, FHID, NN);
  gemm_bf16_kernel<1, true><<<dim3(FHID / BN, NN / BM, NG), 256, 0, stream>>>(
      Y, sH, FHID, Wt2, 0, FHID, b2, Ht0, nullptr, sH, NN, FHID);

  // ---- layer 3 (final output f32 row-major straight to d_out) ----
  gemm_bf16_kernel<0, false><<<dim3(FHID / BN, NN / BM, NG), 256, 0, stream>>>(
      adjn, sAdj, NN, Ht0, sH, NN, nullptr, Y, nullptr, sH, FHID, NN);
  gemm_bf16_kernel<2, true><<<dim3(FOUT / BN, NN / BM, NG), 256, 0, stream>>>(
      Y, sH, FHID, Wt3, 0, FHID, b3, nullptr, out, (size_t)NN * FOUT, FOUT, FHID);
}
